// GATRGCN_37460704755823
// MI455X (gfx1250) — matrix operand order repaired and checked by reading, not compiled, not run on hardware
//
#include <hip/hip_runtime.h>

#define N_NODES 100000
#define N_EDGES 800000
#define NEG_SLOPE 0.2f

typedef __attribute__((ext_vector_type(16))) __bf16 v16bf;
typedef __attribute__((ext_vector_type(8)))  float  v8f;
typedef __attribute__((ext_vector_type(4)))  int    v4i;

#if __has_builtin(__builtin_amdgcn_global_load_async_to_lds_b128) && \
    __has_builtin(__builtin_amdgcn_s_wait_asynccnt)
#define USE_ASYNC_LDS 1
#else
#define USE_ASYNC_LDS 0
#endif

// ---------- helpers ----------
__device__ __forceinline__ unsigned short f2bf(float f) {
  unsigned u = __float_as_uint(f);
  u += 0x7FFFu + ((u >> 16) & 1u);          // round-to-nearest-even
  return (unsigned short)(u >> 16);
}
// monotone float<->uint mapping so unsigned atomicMax == float max
__device__ __forceinline__ unsigned f_ord(float f) {
  unsigned u = __float_as_uint(f);
  return (u & 0x80000000u) ? ~u : (u | 0x80000000u);
}
__device__ __forceinline__ float ord_f(unsigned k) {
  unsigned u = (k & 0x80000000u) ? (k & 0x7FFFFFFFu) : ~k;
  return __uint_as_float(u);
}

// ---------- small utility kernels ----------
__global__ void fill_u32(unsigned* p, unsigned v, int n) {
  int i = blockIdx.x * blockDim.x + threadIdx.x;
  if (i < n) p[i] = v;
}

// n must be a multiple of 4 (always true here: N*128, etc.)
__global__ void f32_to_bf16_x4(const float* __restrict__ in,
                               unsigned short* __restrict__ out, int n4) {
  int i = blockIdx.x * blockDim.x + threadIdx.x;
  if (i >= n4) return;
  float4 f = ((const float4*)in)[i];
  ushort4 o;
  o.x = f2bf(f.x); o.y = f2bf(f.y); o.z = f2bf(f.z); o.w = f2bf(f.w);
  ((ushort4*)out)[i] = o;
}

// Pack W (128x128 f32, row=k, col=n) into per-lane-contiguous WMMA B fragments:
// Bp[((kt*8+ct)*32 + lane)*16 + i] = bf16( W[(kt*32 + (lane>>4)*16 + i)*128 + ct*16 + (lane&15)] )
__global__ void pack_W(const float* __restrict__ W, unsigned short* __restrict__ Bp) {
  int idx = blockIdx.x * blockDim.x + threadIdx.x;   // 16384 total
  if (idx >= 16384) return;
  int i    = idx & 15;
  int lane = (idx >> 4) & 31;
  int ct   = (idx >> 9) & 7;
  int kt   = idx >> 12;
  int k = kt * 32 + (lane >> 4) * 16 + i;
  int n = ct * 16 + (lane & 15);
  Bp[idx] = f2bf(W[k * 128 + n]);
}

// ---------- WMMA GEMM: C(N x 128) = A(N x 128, bf16) * B(128 x 128, packed bf16) ----------
// grid.x = row tiles (N/16); 256 threads = 8 waves; wave w owns column tile w.
// A stripe (16x128 bf16 = 4KB) staged once per block into LDS via async copy,
// then each wave builds fragments with ds_load_b128.
__global__ void __launch_bounds__(256)
gemm_wmma(const unsigned short* __restrict__ A,
          const unsigned short* __restrict__ Bp,
          float* __restrict__ C) {
  __shared__ unsigned short As[16 * 128];

  const int t     = threadIdx.x;
  const int lane  = t & 31;
  const int ct    = t >> 5;                    // col tile 0..7
  const int rt    = blockIdx.x;                // row tile
  const int khalf = lane >> 4;                 // 0 or 1
  const int lrow  = lane & 15;

  // cooperative A-tile copy: 256 threads x 16B = 4KB
  const unsigned short* gsrc = A + (size_t)rt * 16 * 128 + t * 8;
#if USE_ASYNC_LDS
  __builtin_amdgcn_global_load_async_to_lds_b128(
      (__attribute__((address_space(1))) v4i*)gsrc,
      (__attribute__((address_space(3))) v4i*)&As[t * 8], 0, 0);
  __builtin_amdgcn_s_wait_asynccnt(0);
#else
  *(uint4*)&As[t * 8] = *(const uint4*)gsrc;
#endif
  __syncthreads();

  union U32B { uint4 q[2]; v16bf v; };
  v8f acc = {};
#pragma unroll
  for (int kt = 0; kt < 4; ++kt) {
    U32B a, b;
    // A fragment from LDS: elems 0..15 = K kt*32 + khalf*16 .. +15 (contiguous)
    const uint4* ap = (const uint4*)&As[lrow * 128 + kt * 32 + khalf * 16];
    a.q[0] = ap[0];
    a.q[1] = ap[1];
    // B fragment: pre-packed contiguous 32B per lane
    const uint4* bp = (const uint4*)(Bp + ((size_t)(kt * 8 + ct) * 32 + lane) * 16);
    b.q[0] = bp[0];
    b.q[1] = bp[1];
    acc = __builtin_amdgcn_wmma_f32_16x16x32_bf16(
        /*neg_a=*/false, a.v, /*neg_b=*/false, b.v,
        /*c_mod=*/(short)0, acc, /*reuse_a=*/false, /*reuse_b=*/false);
  }
  // C/D layout: VGPR v -> row v + 8*khalf, col = lane&15
  float* cp = C + (size_t)(rt * 16 + 8 * khalf) * 128 + ct * 16 + lrow;
#pragma unroll
  for (int v = 0; v < 8; ++v) cp[v * 128] = acc[v];
}

// ---------- attention score projections: el/er = <feat[n,h,:], al/ar[h,:]> ----------
__global__ void attn_lr(const float* __restrict__ feat, const float* __restrict__ al,
                        const float* __restrict__ ar, float* __restrict__ el,
                        float* __restrict__ er, int H, int D) {
  int i = blockIdx.x * blockDim.x + threadIdx.x;     // over N*H
  if (i >= N_NODES * H) return;
  int n = i / H, h = i % H;
  const float* f = feat + (size_t)n * 128 + h * D;
  float sl = 0.f, sr = 0.f;
  for (int d = 0; d < D; ++d) {
    float v = f[d];
    sl += v * al[h * D + d];
    sr += v * ar[h * D + d];
  }
  el[i] = sl;
  er[i] = sr;
}

// ---------- segment max over incoming edges (ordered-uint atomicMax) ----------
__global__ void edge_max(const int* __restrict__ src, const int* __restrict__ dst,
                         const float* __restrict__ el, const float* __restrict__ er,
                         unsigned* __restrict__ mkey, int H) {
  int e = blockIdx.x * blockDim.x + threadIdx.x;
  if (e >= N_EDGES) return;
  int s = src[e], d = dst[e];
  for (int h = 0; h < H; ++h) {
    float v = el[s * H + h] + er[d * H + h];
    v = v > 0.f ? v : NEG_SLOPE * v;
    atomicMax(&mkey[d * H + h], f_ord(v));
  }
}

// decode keys -> float max; non-finite (untouched sentinel) -> 0, per reference
__global__ void m_fix(unsigned* m, int n) {
  int i = blockIdx.x * blockDim.x + threadIdx.x;
  if (i >= n) return;
  float v = ord_f(m[i]);
  if ((__float_as_uint(v) & 0x7F800000u) == 0x7F800000u) v = 0.f;  // !isfinite
  ((float*)m)[i] = v;
}

// ---------- exp(e - m[dst]) and segment sum ----------
__global__ void edge_expsum(const int* __restrict__ src, const int* __restrict__ dst,
                            const float* __restrict__ el, const float* __restrict__ er,
                            const float* __restrict__ m, float* __restrict__ s,
                            float* __restrict__ ex, int H) {
  int e = blockIdx.x * blockDim.x + threadIdx.x;
  if (e >= N_EDGES) return;
  int ss = src[e], dd = dst[e];
  for (int h = 0; h < H; ++h) {
    float v = el[ss * H + h] + er[dd * H + h];
    v = v > 0.f ? v : NEG_SLOPE * v;
    float x = __expf(v - m[dd * H + h]);
    ex[e * H + h] = x;
    atomicAdd(&s[dd * H + h], x);
  }
}

// ---------- alpha-weighted scatter-add: one wave per edge, lane = 4 dims ----------
// Each lane loads a float4 of feat[src] (global_load_b128) and does 4 f32 atomics.
// The 4 dims never straddle a head boundary since D is a multiple of 4.
__global__ void __launch_bounds__(256)
edge_aggregate(const int* __restrict__ src, const int* __restrict__ dst,
               const float* __restrict__ ex, const float* __restrict__ s,
               const float* __restrict__ feat, float* __restrict__ accum,
               int H, int D) {
  int gid  = blockIdx.x * blockDim.x + threadIdx.x;
  int e    = gid >> 5;                      // wave per edge
  if (e >= N_EDGES) return;
  int lane = threadIdx.x & 31;
  int ss = src[e], dd = dst[e];
  int t0 = lane * 4;                        // dims t0..t0+3
  int h  = t0 / D;
  float alpha = ex[e * H + h] / s[dd * H + h];
  float4 f = *(const float4*)(feat + (size_t)ss * 128 + t0);
  float* ap = accum + (size_t)dd * 128 + t0;
  atomicAdd(ap + 0, alpha * f.x);
  atomicAdd(ap + 1, alpha * f.y);
  atomicAdd(ap + 2, alpha * f.z);
  atomicAdd(ap + 3, alpha * f.w);
}

// ---------- finalize: mean over rels + bias-sum; layer1 adds relu -> bf16 ----------
__global__ void finalize1(const float* __restrict__ accum, const float* __restrict__ b,
                          unsigned short* __restrict__ hbf, int n) {
  int i = blockIdx.x * blockDim.x + threadIdx.x;
  if (i >= n) return;
  int c = i & 127;
  float v = (accum[i] + b[c] + b[128 + c]) * 0.5f;
  hbf[i] = f2bf(v > 0.f ? v : 0.f);
}

__global__ void finalize2(const float* __restrict__ accum, const float* __restrict__ b,
                          float* __restrict__ out, int n) {
  int i = blockIdx.x * blockDim.x + threadIdx.x;
  if (i >= n) return;
  int c = i & 127;
  out[i] = (accum[i] + b[c] + b[128 + c]) * 0.5f;
}

// ---------- launcher ----------
extern "C" void kernel_launch(void* const* d_in, const int* in_sizes, int n_in,
                              void* d_out, int out_size, void* d_ws, size_t ws_size,
                              hipStream_t stream) {
  (void)in_sizes; (void)n_in; (void)out_size; (void)ws_size;
  const float* x   = (const float*)d_in[0];
  const float* W1  = (const float*)d_in[1];
  const float* al1 = (const float*)d_in[2];
  const float* ar1 = (const float*)d_in[3];
  const float* b1  = (const float*)d_in[4];
  const float* W2  = (const float*)d_in[5];
  const float* al2 = (const float*)d_in[6];
  const float* ar2 = (const float*)d_in[7];
  const float* b2  = (const float*)d_in[8];
  const int*   src = (const int*)d_in[9];
  const int*   dst = (const int*)d_in[10];

  char* ws = (char*)d_ws;
  size_t off = 0;
  auto take = [&](size_t bytes) -> char* {
    char* p = ws + off;
    off = (off + bytes + 255) & ~(size_t)255;
    return p;
  };
  unsigned short* hbf   = (unsigned short*)take((size_t)N_NODES * 128 * 2);
  unsigned short* Bp    = (unsigned short*)take((size_t)16384 * 2);
  float*          feat  = (float*)take((size_t)N_NODES * 128 * 4);
  float*          accum = (float*)take((size_t)N_NODES * 128 * 4);
  float*          el    = (float*)take((size_t)N_NODES * 2 * 4);
  float*          er    = (float*)take((size_t)N_NODES * 2 * 4);
  unsigned*       mk    = (unsigned*)take((size_t)N_NODES * 2 * 4);
  float*          sden  = (float*)take((size_t)N_NODES * 2 * 4);
  float*          ex    = (float*)take((size_t)N_EDGES * 2 * 4);

  const int NF = N_NODES * 128;
  const int B256 = 256;

  // layer-1 input in bf16 (vectorized x4)
  f32_to_bf16_x4<<<(NF / 4 + B256 - 1) / B256, B256, 0, stream>>>(x, hbf, NF / 4);

  for (int layer = 0; layer < 2; ++layer) {
    const int H = (layer == 0) ? 2 : 1;
    const int D = 128 / H;
    const float* W  = (layer == 0) ? W1 : W2;
    const float* al = (layer == 0) ? al1 : al2;
    const float* ar = (layer == 0) ? ar1 : ar2;
    const float* b  = (layer == 0) ? b1 : b2;
    const int NH = N_NODES * H;

    fill_u32<<<(NF + B256 - 1) / B256, B256, 0, stream>>>((unsigned*)accum, 0u, NF);

    for (int r = 0; r < 2; ++r) {
      const int* sr = src + (size_t)r * N_EDGES;
      const int* dr = dst + (size_t)r * N_EDGES;
      pack_W<<<(16384 + B256 - 1) / B256, B256, 0, stream>>>(W + (size_t)r * 128 * 128, Bp);
      gemm_wmma<<<N_NODES / 16, 256, 0, stream>>>(hbf, Bp, feat);
      attn_lr<<<(NH + B256 - 1) / B256, B256, 0, stream>>>(feat, al + r * 128, ar + r * 128,
                                                           el, er, H, D);
      fill_u32<<<(NH + B256 - 1) / B256, B256, 0, stream>>>(mk, 0u, NH);
      fill_u32<<<(NH + B256 - 1) / B256, B256, 0, stream>>>((unsigned*)sden, 0u, NH);
      edge_max<<<(N_EDGES + B256 - 1) / B256, B256, 0, stream>>>(sr, dr, el, er, mk, H);
      m_fix<<<(NH + B256 - 1) / B256, B256, 0, stream>>>(mk, NH);
      edge_expsum<<<(N_EDGES + B256 - 1) / B256, B256, 0, stream>>>(sr, dr, el, er,
                                                                    (float*)mk, sden, ex, H);
      edge_aggregate<<<(N_EDGES * 32 + B256 - 1) / B256, B256, 0, stream>>>(
          sr, dr, ex, sden, feat, accum, H, D);
    }
    if (layer == 0)
      finalize1<<<(NF + B256 - 1) / B256, B256, 0, stream>>>(accum, b, hbf, NF);
    else
      finalize2<<<(NF + B256 - 1) / B256, B256, 0, stream>>>(accum, b, (float*)d_out, NF);
  }
}
